// VQ_28432683500141
// MI455X (gfx1250) — matrix operand order repaired and checked by reading, not compiled, not run on hardware
//
#include <hip/hip_runtime.h>
#include <hip/hip_bf16.h>

// Problem constants (from reference): x[B][C][T], codebook[K][C]
#define B_ 32
#define C_ 256
#define T_ 2048
#define K_ 1024

#define MT 128        // tokens per workgroup
#define NCHUNK 64     // codebook rows per LDS chunk
#define XPITCH 272    // 256 + 16 halves pad (bank-conflict mitigation)
#define CBPITCH 272   // 544 B row pitch (16B/32B aligned)

typedef __attribute__((ext_vector_type(16))) _Float16 v16h;
typedef __attribute__((ext_vector_type(8)))  _Float16 v8h;
typedef __attribute__((ext_vector_type(8)))  float    v8f;
typedef __attribute__((ext_vector_type(4)))  float    v4f;

// ---- CDNA5 async global->LDS helpers --------------------------------------
__device__ __forceinline__ unsigned lds_off_of(const void* p) {
    // generic -> LDS(addrspace 3) pointer -> 32-bit LDS byte offset
    return (unsigned)(unsigned long long)
        (__attribute__((address_space(3))) const char*)p;
}

__device__ __forceinline__ void async_b128_to_lds(const void* gptr, unsigned lds_byte) {
    // GLOBAL_LOAD_ASYNC_TO_LDS_B128: LDS[vdst] = MEM[vaddr], tracked by ASYNCcnt
    asm volatile("global_load_async_to_lds_b128 %0, %1, off"
                 :: "v"(lds_byte), "v"(gptr) : "memory");
}

__device__ __forceinline__ void wait_asynccnt0() {
#if __has_builtin(__builtin_amdgcn_s_wait_asynccnt)
    __builtin_amdgcn_s_wait_asynccnt(0);
#else
    asm volatile("s_wait_asynccnt 0x0" ::: "memory");
#endif
}

// ---------------------------------------------------------------------------
// Kernel 0: one-shot codebook prep: f32 -> f16 into workspace, plus
// c2h[r] = 0.5*||codebook[r]||^2 (fp32). One wave per codebook row.
// Grid: 128 blocks x 256 threads = 1024 waves.
// ---------------------------------------------------------------------------
__global__ __launch_bounds__(256) void vq_prep_kernel(
    const float* __restrict__ cb, _Float16* __restrict__ cb16,
    float* __restrict__ c2hg)
{
    const int lane = threadIdx.x & 31;
    const int r    = blockIdx.x * 8 + (threadIdx.x >> 5);   // codebook row
    const int e    = lane * 8;                              // 8 elems per lane

    const float* src = cb + (size_t)r * C_ + e;
    v4f a = *(const v4f*)(src);
    v4f b = *(const v4f*)(src + 4);

    v8h h;
    h[0] = (_Float16)a.x; h[1] = (_Float16)a.y;
    h[2] = (_Float16)a.z; h[3] = (_Float16)a.w;
    h[4] = (_Float16)b.x; h[5] = (_Float16)b.y;
    h[6] = (_Float16)b.z; h[7] = (_Float16)b.w;
    *(v8h*)(cb16 + (size_t)r * C_ + e) = h;

    float s = a.x*a.x + a.y*a.y + a.z*a.z + a.w*a.w
            + b.x*b.x + b.y*b.y + b.z*b.z + b.w*b.w;
    #pragma unroll
    for (int m = 16; m >= 1; m >>= 1)
        s += __shfl_xor(s, m, 32);
    if (lane == 0) c2hg[r] = 0.5f * s;
}

// ---------------------------------------------------------------------------
// Kernel 1: per-token argmin over the codebook via f16 WMMA (f32 accumulate),
// with double-buffered async (DMA) codebook chunk streaming into LDS.
// Grid: 512 WGs x 256 threads. WG handles 128 consecutive t for one b.
// ---------------------------------------------------------------------------
__global__ __launch_bounds__(256) void vq_argmin_kernel(
    const float* __restrict__ x, const _Float16* __restrict__ cb16,
    const float* __restrict__ c2hg,
    int* __restrict__ idxOut, float* __restrict__ idxF)
{
    __shared__ __attribute__((aligned(32))) _Float16 xs[MT * XPITCH];
    __shared__ __attribute__((aligned(32))) _Float16 cs[2][NCHUNK * CBPITCH];
    __shared__ float c2h[K_];

    const int tid  = threadIdx.x;
    const int wave = tid >> 5;
    const int lane = tid & 31;
    const int wg   = blockIdx.x;
    const int b    = wg >> 4;
    const int t0   = (wg & 15) * MT;

    // --- kick off async DMA of chunk 0 immediately (overlaps x-tile load) ---
    {
        const int r  = tid >> 2;
        const int g0 = (tid & 3) * 16;                      // 16B granule column
        const char* src = (const char*)(cb16 + (size_t)r * C_) + g0;
        unsigned    dst = lds_off_of(&cs[0][r * CBPITCH]) + g0;
        #pragma unroll
        for (int g = 0; g < 8; ++g)                         // 8 x 16B = 512B row / 4 lanes
            async_b128_to_lds(src + g * 64, dst + g * 64);
    }

    // --- copy precomputed 0.5*||c||^2 table (4KB) into LDS ---
    {
        v4f v = *(const v4f*)(c2hg + tid * 4);
        *(v4f*)(&c2h[tid * 4]) = v;
    }

    // --- load X tile (coalesced along t), convert to f16, store [m][c] ---
    for (int p = 0; p < 128; ++p) {
        int c = 2 * p + (tid >> 7);
        int m = tid & 127;
        float v = x[((size_t)b * C_ + c) * T_ + (t0 + m)];
        xs[m * XPITCH + c] = (_Float16)v;
    }
    wait_asynccnt0();
    __syncthreads();

    // --- cache this wave's A operand (16 rows x 256 K) in registers.
    // A 16x32 layout (contiguous, matches B): lane<16 -> K=c0..c0+15;
    // lane>=16 -> K=c0+16..c0+31.
    const int mrow = wave * 16 + (lane & 15);
    const int aoff = (lane < 16) ? 0 : 16;
    v16h areg[8];
    #pragma unroll
    for (int kb = 0; kb < 8; ++kb) {
        const _Float16* ab = &xs[mrow * XPITCH + kb * 32 + aoff];
        union { v16h v; v8h h[2]; } u;
        u.h[0] = *(const v8h*)(ab);
        u.h[1] = *(const v8h*)(ab + 8);
        areg[kb] = u.v;
    }

    float bestD[8];
    int   bestI[8];
    #pragma unroll
    for (int v = 0; v < 8; ++v) { bestD[v] = 3.0e38f; bestI[v] = 0; }

    const int col  = lane & 15;            // C/D: lane holds column n%16
    const int boff = (lane < 16) ? 0 : 16; // B: lane>=16 holds K=c0+16..c0+31

    for (int chunk = 0; chunk < K_ / NCHUNK; ++chunk) {
        // async-prefetch next chunk into the other buffer while we compute
        if (chunk + 1 < K_ / NCHUNK) {
            const int n0n = (chunk + 1) * NCHUNK;
            const int r   = tid >> 2;
            const int g0  = (tid & 3) * 16;
            const char* src = (const char*)(cb16 + (size_t)(n0n + r) * C_) + g0;
            unsigned    dst = lds_off_of(&cs[(chunk + 1) & 1][r * CBPITCH]) + g0;
            #pragma unroll
            for (int g = 0; g < 8; ++g)
                async_b128_to_lds(src + g * 64, dst + g * 64);
        }

        const _Float16* cbuf = cs[chunk & 1];
        const int n0 = chunk * NCHUNK;

        #pragma unroll
        for (int nsub = 0; nsub < 4; ++nsub) {
            const int nrow = nsub * 16 + col;
            const _Float16* bb = cbuf + nrow * CBPITCH + boff;
            v8f acc = {};
            #pragma unroll
            for (int kb = 0; kb < 8; ++kb) {
                v16h bv = *(const v16h*)(bb + kb * 32);
                acc = __builtin_amdgcn_wmma_f32_16x16x32_f16(
                    /*neg_a=*/false, areg[kb], /*neg_b=*/false, bv,
                    /*c_mod=*/(short)0, acc, /*reuse_a=*/false, /*reuse_b=*/false);
            }
            // distance surrogate d = 0.5*||c||^2 - x.c ; argmin-equivalent
            const int   n  = n0 + nsub * 16 + col;
            const float ch = c2h[n];
            #pragma unroll
            for (int v = 0; v < 8; ++v) {
                float d = ch - acc[v];
                if (d < bestD[v]) { bestD[v] = d; bestI[v] = n; } // strict <: keeps lowest n
            }
        }

        wait_asynccnt0();   // our prefetch landed (cheap: hidden under 32 wmma)
        __syncthreads();    // everyone done computing old buffer + DMA visible
    }

    // --- cross-lane argmin within each 16-lane half (tie-break: lower index) ---
    #pragma unroll
    for (int v = 0; v < 8; ++v) {
        float d = bestD[v];
        int   i = bestI[v];
        #pragma unroll
        for (int m = 8; m >= 1; m >>= 1) {
            float od = __shfl_xor(d, m, 32);
            int   oi = __shfl_xor(i, m, 32);
            if (od < d || (od == d && oi < i)) { d = od; i = oi; }
        }
        if ((lane & 15) == 0) {
            // C/D layout: lane0 -> row v, lane16 -> row v+8
            int mloc  = wave * 16 + v + ((lane >> 4) << 3);
            int token = b * T_ + t0 + mloc;   // == flat index into [B,1,T]
            idxOut[token] = i;
            idxF[token]   = (float)i;
        }
    }
}

// ---------------------------------------------------------------------------
// Kernel 2: codes[b][c][t] = codebook[idx[b*T+t]][c]
// Thread owns one t, loops c: 64MB of writes fully coalesced along t;
// codebook reads are scattered 16B but the 1MB table is L2-resident.
// ---------------------------------------------------------------------------
__global__ __launch_bounds__(256) void vq_gather_kernel(
    const float* __restrict__ cb, const int* __restrict__ idx,
    float* __restrict__ codes)
{
    const int tid = threadIdx.x;
    const int b   = blockIdx.x >> 3;
    const int t   = ((blockIdx.x & 7) << 8) + tid;

    const int q = idx[b * T_ + t];
    const float* row = cb + (size_t)q * C_;
    float* dst = codes + (size_t)b * C_ * T_ + t;

    #pragma unroll 4
    for (int c = 0; c < C_; c += 4) {
        v4f v = *(const v4f*)(row + c);
        dst[(size_t)(c + 0) * T_] = v.x;
        dst[(size_t)(c + 1) * T_] = v.y;
        dst[(size_t)(c + 2) * T_] = v.z;
        dst[(size_t)(c + 3) * T_] = v.w;
    }
}

extern "C" void kernel_launch(void* const* d_in, const int* in_sizes, int n_in,
                              void* d_out, int out_size, void* d_ws, size_t ws_size,
                              hipStream_t stream) {
    (void)in_sizes; (void)n_in; (void)out_size; (void)ws_size;
    const float* x  = (const float*)d_in[0];   // [B, C, T] fp32
    const float* cb = (const float*)d_in[1];   // [K, C]    fp32

    float* out   = (float*)d_out;
    float* codes = out;                                  // [B, C, T]
    float* idxF  = out + (size_t)B_ * C_ * T_;           // [B, 1, T] as float

    // workspace layout: [0,512K) f16 codebook ; [512K,516K) c2h ; [516K,772K) idx
    _Float16* cb16 = (_Float16*)d_ws;
    float*    c2hg = (float*)((char*)d_ws + (size_t)K_ * C_ * sizeof(_Float16));
    int*      idxWs = (int*)((char*)c2hg + (size_t)K_ * sizeof(float));

    vq_prep_kernel  <<<dim3(128), dim3(256), 0, stream>>>(cb, cb16, c2hg);
    vq_argmin_kernel<<<dim3(512), dim3(256), 0, stream>>>(x, cb16, c2hg, idxWs, idxF);
    vq_gather_kernel<<<dim3(256), dim3(256), 0, stream>>>(cb, idxWs, codes);
}
